// SqueezeExciteBlock_30021821399413
// MI455X (gfx1250) — matrix operand order repaired and checked by reading, not compiled, not run on hardware
//
#include <hip/hip_runtime.h>
#include <hip/hip_fp16.h>

typedef __attribute__((ext_vector_type(16))) _Float16 v16h;
typedef __attribute__((ext_vector_type(8)))  float    v8f;
typedef __attribute__((ext_vector_type(4)))  float    f4;

#define B_      32
#define H_      112
#define W_      112
#define C_      256
#define HW_     (H_ * W_)          // 12544
#define NCHUNK  56
#define CHUNKLEN (HW_ / NCHUNK)    // 224
#define HID     16

// ---------------------------------------------------------------------------
// Kernel 1: partial global-average pool. One block = (batch, spatial chunk),
// thread t = channel t. Wave reads 32 consecutive fp32 channels -> 128B
// contiguous per load, fully coalesced NHWC streaming. 4 accumulators to
// keep 4 loads in flight. Regular (allocating) loads on purpose: the tail of
// this stream stays resident in the 192MB L2 for kernel 3 to hit.
// ---------------------------------------------------------------------------
__global__ __launch_bounds__(256) void se_pool_partial(
    const float* __restrict__ in, float* __restrict__ partial) {
  const int b     = blockIdx.x / NCHUNK;
  const int chunk = blockIdx.x % NCHUNK;
  const int c     = threadIdx.x;

  const float* p = in + (size_t)b * HW_ * C_ + (size_t)chunk * CHUNKLEN * C_ + c;
  float s0 = 0.f, s1 = 0.f, s2 = 0.f, s3 = 0.f;
  for (int i = 0; i < CHUNKLEN; i += 4) {
    s0 += p[(size_t)(i + 0) * C_];
    s1 += p[(size_t)(i + 1) * C_];
    s2 += p[(size_t)(i + 2) * C_];
    s3 += p[(size_t)(i + 3) * C_];
  }
  partial[(size_t)blockIdx.x * C_ + c] = (s0 + s1) + (s2 + s3);
}

// ---------------------------------------------------------------------------
// Kernel 2: finish the mean, then the bottleneck MLP on WMMA (f16 in, f32 acc).
// One workgroup; wave 0 (full EXEC) performs the matrix ops.
//
// VGPR fragment layouts (CDNA5 ISA 7.12.2, wave32, 16-bit data):
//   A 16x32: lane L (g=L/16, m=L%16): element j -> K = j + 8*g + (j>=8 ? 8 : 0)
//   B 32x16: lane L (g=L/16, n=L%16): element j -> K = j + 8*g + (j>=8 ? 8 : 0)
//   C/D 16x16 f32: lane L (g, n): vgpr r -> M = r + 8*g
// ---------------------------------------------------------------------------
__global__ __launch_bounds__(256) void se_mlp_wmma(
    const float* __restrict__ partial,
    const float* __restrict__ w1, const float* __restrict__ b1,
    const float* __restrict__ w2, const float* __restrict__ b2,
    float* __restrict__ gate) {
  __shared__ _Float16 seh[B_][C_];    // squeezed activations [batch][chan]
  __shared__ _Float16 w1h[C_][HID];   // B-matrix for GEMM1 [K][N]
  __shared__ _Float16 w2h[32][C_];    // B-matrix for GEMM2, K padded 16->32
  __shared__ _Float16 hid[B_][32];    // hidden acts, K padded 16->32

  const int t = threadIdx.x;

  // Finish the mean: thread t = channel, loop batches & chunks (coalesced in c).
  for (int b = 0; b < B_; ++b) {
    float s = 0.f;
    const float* pp = partial + (size_t)b * NCHUNK * C_ + t;
    for (int k = 0; k < NCHUNK; ++k) s += pp[(size_t)k * C_];
    seh[b][t] = (_Float16)(s * (1.0f / (float)HW_));
  }
  // Stage weights as f16.
  for (int i = t; i < C_ * HID; i += 256) w1h[i / HID][i % HID] = (_Float16)w1[i];
  for (int i = t; i < 16 * C_;  i += 256) w2h[i / C_][i % C_]   = (_Float16)w2[i];
  for (int i = t; i < 16 * C_;  i += 256) w2h[16 + i / C_][i % C_] = (_Float16)0.0f;
  for (int i = t; i < B_ * 32;  i += 256) hid[i / 32][i % 32]   = (_Float16)0.0f;
  __syncthreads();

  if (t < 32) {           // wave 0 only: EXEC is all ones inside this wave
    const int lane = t;
    const int g  = lane >> 4;    // half-wave group
    const int mn = lane & 15;    // M for A-frags, N for B/C/D-frags

    // ---- GEMM1: se[32,256] x w1[256,16] -> hidden[32,16], relu(+b1) ----
    for (int mt = 0; mt < 2; ++mt) {
      v8f acc = {};
      for (int k0 = 0; k0 < C_; k0 += 32) {
        v16h a, bb;
#pragma unroll
        for (int j = 0; j < 16; ++j) {
          const int ka = k0 + j + 8 * g + ((j >= 8) ? 8 : 0);
          a[j] = seh[mt * 16 + mn][ka];
          const int kb = k0 + j + 8 * g + ((j >= 8) ? 8 : 0);
          bb[j] = w1h[kb][mn];
        }
        acc = __builtin_amdgcn_wmma_f32_16x16x32_f16(
            false, a, false, bb, (short)0, acc, false, false);
      }
      const float bias = b1[mn];
#pragma unroll
      for (int r = 0; r < 8; ++r) {
        const float v = acc[r] + bias;
        hid[mt * 16 + r + 8 * g][mn] = (_Float16)(v > 0.f ? v : 0.f);
      }
    }

    // ---- GEMM2: hidden[32,16(pad32)] x w2[16(pad32),256] -> gate, sigmoid(+b2)
    for (int mt = 0; mt < 2; ++mt) {
      v16h a;
#pragma unroll
      for (int j = 0; j < 16; ++j) {
        const int ka = j + 8 * g + ((j >= 8) ? 8 : 0);
        a[j] = hid[mt * 16 + mn][ka];
      }
      for (int nt = 0; nt < 16; ++nt) {
        v16h bb;
#pragma unroll
        for (int j = 0; j < 16; ++j) {
          const int kb = j + 8 * g + ((j >= 8) ? 8 : 0);
          bb[j] = w2h[kb][nt * 16 + mn];
        }
        v8f acc = {};
        acc = __builtin_amdgcn_wmma_f32_16x16x32_f16(
            false, a, false, bb, (short)0, acc, false, false);
        const int col  = nt * 16 + mn;
        const float bias = b2[col];
#pragma unroll
        for (int r = 0; r < 8; ++r) {
          const float x = acc[r] + bias;
          const float s = 1.0f / (1.0f + __expf(-x));
          gate[(size_t)(mt * 16 + r + 8 * g) * C_ + col] = s;
        }
      }
    }
  }
}

// ---------------------------------------------------------------------------
// Kernel 3: broadcast-multiply, float4 streaming.
//  - Reverse block order: first blocks touch the tensor tail, which is still
//    resident in the 192MB L2 from kernel 1's front-to-back stream.
//  - Non-temporal loads/stores: this traffic is dead after use; don't let it
//    evict the reusable resident window. gate (32KB) stays cache-hot.
// ---------------------------------------------------------------------------
__global__ __launch_bounds__(256) void se_scale(
    const f4* __restrict__ in, const float* __restrict__ gate,
    f4* __restrict__ out, int n4) {
  const int blk = (int)gridDim.x - 1 - (int)blockIdx.x;   // reverse traversal
  const int i   = blk * 256 + (int)threadIdx.x;
  if (i >= n4) return;
  const size_t flat = (size_t)i * 4;
  const int c = (int)(flat & (C_ - 1));                   // C_=256, 4 | flat
  const int b = (int)(flat / ((size_t)HW_ * C_));
  const float g0 = gate[(size_t)b * C_ + c + 0];
  const float g1 = gate[(size_t)b * C_ + c + 1];
  const float g2 = gate[(size_t)b * C_ + c + 2];
  const float g3 = gate[(size_t)b * C_ + c + 3];
  f4 v = __builtin_nontemporal_load(&in[i]);
  v.x *= g0; v.y *= g1; v.z *= g2; v.w *= g3;
  __builtin_nontemporal_store(v, &out[i]);
}

extern "C" void kernel_launch(void* const* d_in, const int* in_sizes, int n_in,
                              void* d_out, int out_size, void* d_ws, size_t ws_size,
                              hipStream_t stream) {
  const float* in = (const float*)d_in[0];
  const float* w1 = (const float*)d_in[1];
  const float* b1 = (const float*)d_in[2];
  const float* w2 = (const float*)d_in[3];
  const float* b2 = (const float*)d_in[4];
  float* out = (float*)d_out;

  float* partial = (float*)d_ws;                          // 32*56*256 fp32
  float* gate    = partial + (size_t)B_ * NCHUNK * C_;    // 32*256 fp32

  se_pool_partial<<<B_ * NCHUNK, 256, 0, stream>>>(in, partial);
  se_mlp_wmma<<<1, 256, 0, stream>>>(partial, w1, b1, w2, b2, gate);

  const int n4 = (B_ * HW_ * C_) / 4;
  se_scale<<<(n4 + 255) / 256, 256, 0, stream>>>(
      (const f4*)in, gate, (f4*)out, n4);
}
